// Attention_ASMD_15315853377993
// MI455X (gfx1250) — matrix operand order repaired and checked by reading, not compiled, not run on hardware
//
#include <hip/hip_runtime.h>
#include <stdint.h>

// ---------------------------------------------------------------------------
// MI455X (gfx1250) implementation of the spectral-attention block.
//
// Pipeline (bandwidth-bound; bf16 WMMA 16x16x32 w/ f32 accum for GEMMs):
//   1. k_prep    : w_qkv f32->bf16, zero Gram/norm accumulators
//   2. k_cvt_x   : x f32->bf16 (NT loads; single-use stream)
//   3. k_gemm_n  : qkv0 = Wqkv @ x        (M=576, K=192, N=65536, per batch)
//   4. k_dw      : 3x3 depthwise + fused sum-of-squares for q/k rows
//   5. k_gram    : S = q @ k^T  (split-K over 64 chunks, f32 atomics)
//   6. k_spec    : normalize Gram, 48x48 DFT, 4 masked |iDFT|, weighted sum
//   7. k_meff    : Meff = Wproj @ blockdiag(Aeff)   (192x192 per batch)
//   8. k_gemm_n  : out = Meff @ v          (M=192, K=192, N=65536, NT stores)
//
// k_gemm_n uses double-buffered global_load_async_to_lds_b128 staging and
// reuses each LDS B-fragment across two m-subtiles (2 WMMA per fragment).
// ---------------------------------------------------------------------------

typedef unsigned short u16;
typedef __attribute__((ext_vector_type(16))) __bf16        v16bf;
typedef __attribute__((ext_vector_type(8)))  float         v8f;
typedef __attribute__((ext_vector_type(4)))  unsigned int  u32x4;
typedef __attribute__((ext_vector_type(4)))  float         f32x4;
typedef __attribute__((ext_vector_type(4)))  u16           u16x4;

#define NPIX   65536
#define WIMG   256
#define CDIM   192
#define C3     576
#define NHEAD  4
#define CH     48
#define NB     2

#define USE_ASYNC_LDS 1
#define TPITCH 136                               // LDS row pitch (bf16 units)

__device__ __forceinline__ u16 f2bf(float f) {
  unsigned u = __float_as_uint(f);
  return (u16)((u + 0x7FFFu + ((u >> 16) & 1u)) >> 16);   // RNE
}
__device__ __forceinline__ float bf2f(u16 s) {
  return __uint_as_float(((unsigned)s) << 16);
}

__device__ __forceinline__ void async_b128(const u16* src, u16* dst_lds) {
#if USE_ASYNC_LDS
  unsigned ldsoff = (unsigned)(size_t)dst_lds;   // addr[31:0] = LDS offset
  asm volatile("global_load_async_to_lds_b128 %0, %1, off"
               :: "v"(ldsoff), "v"(src) : "memory");
#else
  *(u32x4*)dst_lds = *(const u32x4*)src;
#endif
}
__device__ __forceinline__ void wait_async() {
#if USE_ASYNC_LDS
#if __has_builtin(__builtin_amdgcn_s_wait_asynccnt)
  __builtin_amdgcn_s_wait_asynccnt(0);
#else
  asm volatile("s_wait_asynccnt 0" ::: "memory");
#endif
#endif
}

union ABFrag { u32x4 u4[2]; u16 s[16]; v16bf v; };

// ---------------------------------------------------------------------------
// 1. weight convert + zero accumulators (SG, NQ, NK are contiguous in ws)
// ---------------------------------------------------------------------------
__global__ __launch_bounds__(256)
void k_prep(const float* __restrict__ wqkv, u16* __restrict__ WB,
            float* __restrict__ zr) {
  int idx = blockIdx.x * 256 + threadIdx.x;
  if (idx < C3 * CDIM) WB[idx] = f2bf(wqkv[idx]);
  if (idx < 19200)     zr[idx] = 0.0f;
}

// ---------------------------------------------------------------------------
// 2. x f32 -> bf16 (non-temporal loads: x is read exactly once)
// ---------------------------------------------------------------------------
__global__ __launch_bounds__(256)
void k_cvt_x(const float* __restrict__ x, u16* __restrict__ XB) {
  size_t i = ((size_t)blockIdx.x * 256 + threadIdx.x) * 4;
  f32x4 f = __builtin_nontemporal_load((const f32x4*)(x + i));
  u16x4 o = { f2bf(f[0]), f2bf(f[1]), f2bf(f[2]), f2bf(f[3]) };
  *(u16x4*)(XB + i) = o;
}

// ---------------------------------------------------------------------------
// 3/8. Streaming GEMM: C[M x NPIX] = A[M x 192] @ B[192 x NPIX] per batch.
//      Block 256 thr = 8 waves covering 32 rows x 128 cols. B tiles (32x128)
//      double-buffered in LDS via async global->LDS; each wave owns one
//      16-col strip and two 16-row subtiles (B fragment reused, 2 WMMA).
// ---------------------------------------------------------------------------
template<bool STORE_BF16>
__global__ __launch_bounds__(256)
void k_gemm_n(const u16* __restrict__ A, size_t a_bstride,
              const u16* __restrict__ Bm, size_t b_bstride,
              void* __restrict__ Cout, int M) {
  __shared__ u16 tile[2][32][TPITCH];
  const int n0 = blockIdx.x * 128;
  const int m0 = blockIdx.y * 32;
  const int b  = blockIdx.z;
  const u16* __restrict__ Ab = A  + (size_t)b * a_bstride;
  const u16* __restrict__ Bb = Bm + (size_t)b * b_bstride;

  const int t    = threadIdx.x;
  const int lane = t & 31;
  const int nsub = t >> 5;                     // wave id = column strip 0..7
  const int lo   = lane & 15;
  const int hi   = lane >> 4;
  const int srow = t >> 3;                     // staging row 0..31
  const int scol = (t & 7) * 8;                // staging col (bf16 units)

  v8f acc0 = {0.f,0.f,0.f,0.f,0.f,0.f,0.f,0.f};
  v8f acc1 = {0.f,0.f,0.f,0.f,0.f,0.f,0.f,0.f};

  // prologue: stage first K-tile into buffer 0 (2 x 16B per thread)
  {
    const u16* src = Bb + (size_t)srow * NPIX + n0 + scol;
    async_b128(src,      &tile[0][srow][scol]);
    async_b128(src + 64, &tile[0][srow][scol + 64]);
  }

  int cur = 0;
  for (int k0 = 0; k0 < CDIM; k0 += 32) {
    wait_async();                              // tile[cur] resident (per-wave)
    __syncthreads();                           // ... for all waves; prev reads done
    if (k0 + 32 < CDIM) {                      // stage next tile into tile[cur^1]
      const u16* src = Bb + (size_t)(k0 + 32 + srow) * NPIX + n0 + scol;
      async_b128(src,      &tile[cur ^ 1][srow][scol]);
      async_b128(src + 64, &tile[cur ^ 1][srow][scol + 64]);
    }

    // B fragment: column (lane&15) of strip nsub, K-half (lane>>4), K-major
    ABFrag bfr;
#pragma unroll
    for (int tt = 0; tt < 16; ++tt)
      bfr.s[tt] = tile[cur][hi * 16 + tt][nsub * 16 + lo];

    // A fragments: row per lane, contiguous 16-elem K-half (lane>>4)
    ABFrag a0, a1;
    const u16* ap0 = Ab + (size_t)(m0 + lo) * CDIM + k0 + hi * 16;
    a0.u4[0] = *(const u32x4*)ap0;
    a0.u4[1] = *(const u32x4*)(ap0 + 8);
    const u16* ap1 = ap0 + 16 * CDIM;
    a1.u4[0] = *(const u32x4*)ap1;
    a1.u4[1] = *(const u32x4*)(ap1 + 8);

    acc0 = __builtin_amdgcn_wmma_f32_16x16x32_bf16(false, a0.v, false, bfr.v,
                                                   (short)0, acc0, false, false);
    acc1 = __builtin_amdgcn_wmma_f32_16x16x32_bf16(false, a1.v, false, bfr.v,
                                                   (short)0, acc1, false, false);
    cur ^= 1;
  }

  const int col = n0 + nsub * 16 + lo;
  const int r0  = m0 + (lane < 16 ? 0 : 8);
  if (STORE_BF16) {
    u16* C = (u16*)Cout + (size_t)b * M * (size_t)NPIX;
#pragma unroll
    for (int vi = 0; vi < 8; ++vi) {
      C[(size_t)(r0 + vi) * NPIX + col]      = f2bf(acc0[vi]);
      C[(size_t)(r0 + 16 + vi) * NPIX + col] = f2bf(acc1[vi]);
    }
  } else {
    float* C = (float*)Cout + (size_t)b * M * (size_t)NPIX;
#pragma unroll
    for (int vi = 0; vi < 8; ++vi) {          // NT: output is never re-read
      __builtin_nontemporal_store(acc0[vi], &C[(size_t)(r0 + vi) * NPIX + col]);
      __builtin_nontemporal_store(acc1[vi], &C[(size_t)(r0 + 16 + vi) * NPIX + col]);
    }
  }
}

// ---------------------------------------------------------------------------
// 4. 3x3 depthwise (zero pad) + fused q/k row sum-of-squares
// ---------------------------------------------------------------------------
__global__ __launch_bounds__(256)
void k_dw(const u16* __restrict__ qkv0, const float* __restrict__ wdw,
          u16* __restrict__ qkv1, float* __restrict__ nq, float* __restrict__ nk) {
  const int x = threadIdx.x, y = blockIdx.x, ch = blockIdx.y, b = blockIdx.z;
  const size_t pbase = ((size_t)b * C3 + ch) * (size_t)NPIX;
  const u16* P = qkv0 + pbase;
  float wv[9];
#pragma unroll
  for (int i = 0; i < 9; ++i) wv[i] = wdw[ch * 9 + i];   // uniform -> SMEM
  float s = 0.f;
#pragma unroll
  for (int dy = -1; dy <= 1; ++dy) {
    int yy = y + dy;
    if (yy < 0 || yy >= WIMG) continue;
#pragma unroll
    for (int dx = -1; dx <= 1; ++dx) {
      int xx = x + dx;
      if (xx < 0 || xx >= WIMG) continue;
      s += wv[(dy + 1) * 3 + (dx + 1)] * bf2f(P[yy * WIMG + xx]);
    }
  }
  qkv1[pbase + y * WIMG + x] = f2bf(s);

  if (ch < 2 * CDIM) {                         // q or k channel: accumulate ||row||^2
    float v2 = s * s;
#pragma unroll
    for (int off = 16; off > 0; off >>= 1)
      v2 += __shfl_down(v2, off, 32);
    __shared__ float wsum[8];
    int lane = threadIdx.x & 31, wave = threadIdx.x >> 5;
    if (lane == 0) wsum[wave] = v2;
    __syncthreads();
    if (threadIdx.x == 0) {
      float tot = 0.f;
      for (int i = 0; i < 8; ++i) tot += wsum[i];
      int cc = (ch < CDIM) ? ch : ch - CDIM;
      float* dst = (ch < CDIM) ? nq : nk;
      atomicAdd(&dst[(b * NHEAD + cc / CH) * CH + cc % CH], tot);
    }
  }
}

// ---------------------------------------------------------------------------
// 5. Gram: S[bh][48][48] += q @ k^T over K-chunk (split-K=64, f32 atomics).
//    Both operands are n-contiguous -> direct per-lane 128-bit loads.
// ---------------------------------------------------------------------------
__global__ __launch_bounds__(288)
void k_gram(const u16* __restrict__ qkv1, float* __restrict__ S) {
  const int chunk = blockIdx.x;                // 64 K-chunks of 1024
  const int bh = blockIdx.y;
  const int b = bh >> 2, h = bh & 3;
  const int wave = threadIdx.x >> 5;           // 9 waves -> 3x3 output tiles
  const int lane = threadIdx.x & 31;
  const int ti = wave / 3, tj = wave - ti * 3;
  const int lo = lane & 15, hi = lane >> 4;
  const u16* qrow = qkv1 + ((size_t)b * C3 + h * CH + ti * 16 + lo) * (size_t)NPIX;
  const u16* krow = qkv1 + ((size_t)b * C3 + CDIM + h * CH + tj * 16 + lo) * (size_t)NPIX;
  v8f acc = {0.f,0.f,0.f,0.f,0.f,0.f,0.f,0.f};
  const int kbeg = chunk * (NPIX / 64);
  for (int k0 = kbeg; k0 < kbeg + NPIX / 64; k0 += 32) {
    ABFrag a, bb;
    const u16* ap = qrow + k0 + hi * 16;       // A: q rows
    a.u4[0] = *(const u32x4*)ap;
    a.u4[1] = *(const u32x4*)(ap + 8);
    const u16* bp = krow + k0 + hi * 16;       // B: k^T columns = k rows
    bb.u4[0] = *(const u32x4*)bp;
    bb.u4[1] = *(const u32x4*)(bp + 8);
    acc = __builtin_amdgcn_wmma_f32_16x16x32_bf16(false, a.v, false, bb.v,
                                                  (short)0, acc, false, false);
  }
  float* Sp = S + bh * (CH * CH);
  const int r0 = ti * 16 + (lane < 16 ? 0 : 8);
  const int col = tj * 16 + lo;
#pragma unroll
  for (int vi = 0; vi < 8; ++vi)
    atomicAdd(&Sp[(r0 + vi) * CH + col], acc[vi]);
}

// ---------------------------------------------------------------------------
// 6. Spectral stage per (b,head): normalize Gram, separable 48x48 DFT
//    (norm='forward'), 4 high-pass masks, |iDFT|, weighted sum -> Aeff.
// ---------------------------------------------------------------------------
__global__ __launch_bounds__(256)
void k_spec(const float* __restrict__ S, const float* __restrict__ nq,
            const float* __restrict__ nk, const float* __restrict__ temp,
            const float* __restrict__ a1, const float* __restrict__ a2,
            const float* __restrict__ a3, const float* __restrict__ a4,
            float* __restrict__ Aeff) {
  __shared__ float Am[CH][CH], Wc[CH][CH], Wsn[CH][CH];
  __shared__ float T1r[CH][CH], T1i[CH][CH], Fr[CH][CH], Fi[CH][CH], Acc[CH][CH];
  const int bh = blockIdx.x, h = bh & 3;

  for (int idx = threadIdx.x; idx < CH * CH; idx += 256) {
    int u = idx / CH, j = idx % CH;
    float ang = 6.283185307179586f * (float)((u * j) % CH) / (float)CH;
    Wc[u][j] = __cosf(ang);
    Wsn[u][j] = __sinf(ang);
  }
  for (int idx = threadIdx.x; idx < CH * CH; idx += 256) {
    int c = idx / CH, d = idx % CH;
    float qn = fmaxf(sqrtf(nq[bh * CH + c]), 1e-12f);
    float kn = fmaxf(sqrtf(nk[bh * CH + d]), 1e-12f);
    Am[c][d] = S[bh * CH * CH + idx] / (qn * kn) * temp[h];
    Acc[c][d] = 0.f;
  }
  __syncthreads();
  // T1 = W @ Am  (W = exp(-i 2pi u c / 48))
  for (int idx = threadIdx.x; idx < CH * CH; idx += 256) {
    int u = idx / CH, d = idx % CH;
    float sr = 0.f, si = 0.f;
    for (int c = 0; c < CH; ++c) {
      float av = Am[c][d];
      sr += Wc[u][c] * av;
      si -= Wsn[u][c] * av;
    }
    T1r[u][d] = sr; T1i[u][d] = si;
  }
  __syncthreads();
  const float inv = 1.0f / (float)(CH * CH);   // norm='forward'
  for (int idx = threadIdx.x; idx < CH * CH; idx += 256) {
    int u = idx / CH, v = idx % CH;
    float fr = 0.f, fi = 0.f;
    for (int d = 0; d < CH; ++d) {
      float tr = T1r[u][d], tiv = T1i[u][d];
      float cw = Wc[v][d], sw = Wsn[v][d];
      fr += tr * cw + tiv * sw;
      fi += tiv * cw - tr * sw;
    }
    Fr[u][v] = fr * inv; Fi[u][v] = fi * inv;
  }
  const float wgt[4] = { a1[0], a2[0], a3[0], a4[0] };
  const int   hhv[4] = { 4, 9, 14, 19 };       // int(48*ratio)
  for (int br = 0; br < 4; ++br) {
    const int hh = hhv[br];
    __syncthreads();
    // T2[c][v] = sum_u g(u,v) F[u][v] e^{+i 2pi u c / 48}  (reuse T1 storage)
    for (int idx = threadIdx.x; idx < CH * CH; idx += 256) {
      int c = idx / CH, v = idx % CH;
      int vr = (v + CH / 2) % CH;
      bool vin = (vr >= CH / 2 - hh) && (vr < CH / 2 + hh);
      float sr = 0.f, si = 0.f;
      for (int u = 0; u < CH; ++u) {
        int ur = (u + CH / 2) % CH;
        bool uin = (ur >= CH / 2 - hh) && (ur < CH / 2 + hh);
        if (uin && vin) continue;              // center box zeroed (high-pass)
        float gr = Fr[u][v], gi = Fi[u][v];
        float cw = Wc[u][c], sw = Wsn[u][c];
        sr += gr * cw - gi * sw;
        si += gr * sw + gi * cw;
      }
      T1r[c][v] = sr; T1i[c][v] = si;
    }
    __syncthreads();
    for (int idx = threadIdx.x; idx < CH * CH; idx += 256) {
      int c = idx / CH, d = idx % CH;
      float ar = 0.f, ai = 0.f;
      for (int v = 0; v < CH; ++v) {
        float tr = T1r[c][v], tiv = T1i[c][v];
        float cw = Wc[v][d], sw = Wsn[v][d];
        ar += tr * cw - tiv * sw;
        ai += tr * sw + tiv * cw;
      }
      Acc[c][d] += wgt[br] * sqrtf(ar * ar + ai * ai);
    }
  }
  __syncthreads();
  for (int idx = threadIdx.x; idx < CH * CH; idx += 256)
    Aeff[bh * CH * CH + idx] = Acc[idx / CH][idx % CH];
}

// ---------------------------------------------------------------------------
// 7. Meff = Wproj @ blockdiag(Aeff) -> bf16 (A-operand of final GEMM)
// ---------------------------------------------------------------------------
__global__ __launch_bounds__(256)
void k_meff(const float* __restrict__ Aeff, const float* __restrict__ wproj,
            u16* __restrict__ MB) {
  const int b = blockIdx.x;
  for (int idx = threadIdx.x; idx < CDIM * CDIM; idx += 256) {
    int o = idx / CDIM, m = idx % CDIM;
    int h = m / CH, d = m % CH;
    const float* wrow = wproj + o * CDIM + h * CH;
    const float* arow = Aeff + (b * NHEAD + h) * CH * CH + d;
    float s = 0.f;
#pragma unroll
    for (int c = 0; c < CH; ++c) s += wrow[c] * arow[c * CH];
    MB[b * CDIM * CDIM + idx] = f2bf(s);
  }
}

// ---------------------------------------------------------------------------
extern "C" void kernel_launch(void* const* d_in, const int* in_sizes, int n_in,
                              void* d_out, int out_size, void* d_ws, size_t ws_size,
                              hipStream_t stream) {
  const float* x     = (const float*)d_in[0];
  const float* wqkv  = (const float*)d_in[1];
  const float* wdw   = (const float*)d_in[2];
  const float* wproj = (const float*)d_in[3];
  const float* temp  = (const float*)d_in[4];
  const float* a1    = (const float*)d_in[5];
  const float* a2    = (const float*)d_in[6];
  const float* a3    = (const float*)d_in[7];
  const float* a4    = (const float*)d_in[8];

  char* ws = (char*)d_ws;
  u16*   WB   = (u16*)(ws + 0);                       // 221184 B
  u16*   MB   = (u16*)(ws + 221184);                  // 147456 B
  float* SG   = (float*)(ws + 368640);                // 73728 B
  float* NQ   = (float*)(ws + 442368);                // 1536 B
  float* NK   = (float*)(ws + 443904);                // 1536 B
  float* AE   = (float*)(ws + 445440);                // 73728 B
  u16*   QKV0 = (u16*)(ws + 519168);                  // 150994944 B
  u16*   QKV1 = (u16*)(ws + 519168 + 150994944ull);   // 150994944 B
  u16*   XB   = QKV1;   // aliased: XB is dead before k_dw writes QKV1

  k_prep<<<432, 256, 0, stream>>>(wqkv, WB, SG);      // SG,NQ,NK contiguous
  k_cvt_x<<<24576, 256, 0, stream>>>(x, XB);
  k_gemm_n<true><<<dim3(512, 18, NB), 256, 0, stream>>>(
      WB, 0, XB, (size_t)CDIM * NPIX, QKV0, C3);
  k_dw<<<dim3(WIMG, C3, NB), 256, 0, stream>>>(QKV0, wdw, QKV1, NQ, NK);
  k_gram<<<dim3(64, NB * NHEAD), 288, 0, stream>>>(QKV1, SG);
  k_spec<<<NB * NHEAD, 256, 0, stream>>>(SG, NQ, NK, temp, a1, a2, a3, a4, AE);
  k_meff<<<NB, 256, 0, stream>>>(AE, wproj, MB);
  k_gemm_n<false><<<dim3(512, 6, NB), 256, 0, stream>>>(
      MB, (size_t)CDIM * CDIM, QKV1 + (size_t)2 * CDIM * NPIX,
      (size_t)C3 * NPIX, d_out, CDIM);
}
